// voMoE_26276609917409
// MI455X (gfx1250) — matrix operand order repaired and checked by reading, not compiled, not run on hardware
//
#include <hip/hip_runtime.h>
#include <hip/hip_fp16.h>
#include <stdint.h>

#define HD   768
#define NE   8
#define NTOK 32768          // B*S = 4*8192
#define TM   32             // tokens per expert-GEMM block (2 WMMA M-tiles)
#define CAP  32768          // max entries per expert list
#define LDA  776            // padded LDS row stride (halfs) -> avoids bank conflicts

typedef __attribute__((ext_vector_type(16))) _Float16 v16h;
typedef __attribute__((ext_vector_type(4)))  _Float16 v4h;
typedef __attribute__((ext_vector_type(8)))  float    v8f;

union Frag { v16h h; uint4 q[2]; };

// ---------------------------------------------------------------------------
// Kernel 1: zero output + per-expert counters
// ---------------------------------------------------------------------------
__global__ __launch_bounds__(256) void moe_zero_kernel(float* __restrict__ out,
                                                       int n, int* __restrict__ counts) {
    int idx    = blockIdx.x * blockDim.x + threadIdx.x;
    int stride = gridDim.x * blockDim.x;
    for (int i = idx; i < n; i += stride) out[i] = 0.0f;
    if (idx < NE) counts[idx] = 0;
}

// ---------------------------------------------------------------------------
// Kernel 2: expert weights f32 -> f16 (once; weights then live in L2 as f16)
// ---------------------------------------------------------------------------
__global__ __launch_bounds__(256) void moe_cvt_kernel(const float* __restrict__ w,
                                                      _Float16* __restrict__ wh, int n4) {
    int idx    = blockIdx.x * blockDim.x + threadIdx.x;
    int stride = gridDim.x * blockDim.x;
    for (int i = idx; i < n4; i += stride) {
        float4 v = ((const float4*)w)[i];
        v4h h;
        h.x = (_Float16)v.x; h.y = (_Float16)v.y;
        h.z = (_Float16)v.z; h.w = (_Float16)v.w;
        ((v4h*)wh)[i] = h;
    }
}

// ---------------------------------------------------------------------------
// Kernel 3: router — one wave32 per token.
// scores = x @ rw^T + rb ; top-2 ; weights = renormalized softmax probs.
// Appends (token, weight) to per-expert gather lists.
// ---------------------------------------------------------------------------
__global__ __launch_bounds__(256) void moe_router_kernel(const float* __restrict__ x,
                                                         const float* __restrict__ rw,
                                                         const float* __restrict__ rb,
                                                         int*  __restrict__ counts,
                                                         int2* __restrict__ entries) {
    __shared__ float RW[NE * HD];                       // 24 KB router weights
    int tid = threadIdx.x;
    for (int i = tid; i < NE * HD; i += 256) RW[i] = rw[i];
    __syncthreads();

    int wave = tid >> 5, lane = tid & 31;
    int t = blockIdx.x * 8 + wave;                      // token id
    const float* xr = x + (size_t)t * HD;

    float acc[NE];
#pragma unroll
    for (int e = 0; e < NE; ++e) acc[e] = 0.0f;

#pragma unroll 4
    for (int i = 0; i < HD / 32; ++i) {
        int h = lane + 32 * i;
        float xv = xr[h];
#pragma unroll
        for (int e = 0; e < NE; ++e) acc[e] += xv * RW[e * HD + h];
    }
    // wave32 reduction
#pragma unroll
    for (int off = 16; off > 0; off >>= 1) {
#pragma unroll
        for (int e = 0; e < NE; ++e) acc[e] += __shfl_xor(acc[e], off, 32);
    }

    if (lane == 0) {
#pragma unroll
        for (int e = 0; e < NE; ++e) acc[e] += rb[e];
        int i1 = 0; float v1 = acc[0];
#pragma unroll
        for (int e = 1; e < NE; ++e) if (acc[e] > v1) { v1 = acc[e]; i1 = e; }
        int i2 = (i1 == 0) ? 1 : 0; float v2 = acc[i2];
#pragma unroll
        for (int e = 0; e < NE; ++e)
            if (e != i1 && acc[e] > v2) { v2 = acc[e]; i2 = e; }
        // w1 = p1/(p1+p2), w2 = p2/(p1+p2) with p = exp(s - s_max)
        float e2 = __expf(v2 - v1);
        float inv = 1.0f / (1.0f + e2);
        float w1 = inv, w2 = e2 * inv;
        int p1 = atomicAdd(&counts[i1], 1);
        entries[i1 * CAP + p1] = make_int2(t, __float_as_int(w1));
        int p2 = atomicAdd(&counts[i2], 1);
        entries[i2 * CAP + p2] = make_int2(t, __float_as_int(w2));
    }
}

// ---------------------------------------------------------------------------
// WMMA fragment loaders (ISA 7.12.2 layouts, 16-bit, wave32)
// A 16x32: lane m=l%16, g=l/16 ; VGPR v<4: K=2v+8g ; v>=4: K=2(v-4)+16+8g
//   => contiguous halfs [k0+8g .. +7] and [k0+16+8g .. +7]  (2x b128)
// B 32x16: lane n=l%16, g=l/16 ; VGPR v<4: K=2v+8g ; v>=4: K=2(v-4)+16+8g
//   => contiguous halfs [k0+8g .. +7] and [k0+16+8g .. +7]                     (2x b128)
// ---------------------------------------------------------------------------
__device__ __forceinline__ void load_a_frag(Frag& f, const _Float16* A,
                                            int mt, int k0, int lane) {
    int m = lane & 15, g = lane >> 4;
    const _Float16* p = A + (mt * 16 + m) * LDA + k0 + 8 * g;
    f.q[0] = *(const uint4*)p;
    f.q[1] = *(const uint4*)(p + 16);
}

__device__ __forceinline__ void load_b_frag(Frag& f, const _Float16* W,
                                            int n0, int k0, int lane) {
    int n = n0 + (lane & 15), g = lane >> 4;
    const _Float16* p = W + (size_t)n * HD + k0 + 8 * g;
    f.q[0] = *(const uint4*)p;
    f.q[1] = *(const uint4*)(p + 16);
}

// ---------------------------------------------------------------------------
// Kernel 4: gathered expert GEMM.
// Block = (tile of TM=32 tokens of expert e) x (full N=768).
// 8 waves; each wave owns 6 N-tiles x 2 M-tiles = 12 accumulators (v8f).
// ---------------------------------------------------------------------------
__global__ __launch_bounds__(256) void moe_expert_kernel(const float* __restrict__ x,
                                                         const _Float16* __restrict__ Wh,
                                                         const float* __restrict__ eb,
                                                         const int*  __restrict__ counts,
                                                         const int2* __restrict__ entries,
                                                         float* __restrict__ out) {
    int e     = blockIdx.y;
    int start = blockIdx.x * TM;
    int cnt   = counts[e];
    if (start >= cnt) return;                 // block-uniform: EXEC stays full below

    __shared__ _Float16 A[TM * LDA];          // ~48.5 KB gathered f16 activations
    __shared__ int   toks[TM];
    __shared__ float wts[TM];

    int tid = threadIdx.x;
    if (tid < TM) {
        int idx = start + tid;
        if (idx < cnt) {
            int2 en = entries[e * CAP + idx];
            toks[tid] = en.x;
            wts[tid]  = __int_as_float(en.y);
        } else {
            toks[tid] = 0;
            wts[tid]  = 0.0f;                 // zero weight -> contributes 0
        }
    }
    __syncthreads();

    int wave = tid >> 5, lane = tid & 31;

    // gather + convert x rows into LDS (coalesced: one row per wave pass)
    for (int r = wave; r < TM; r += 8) {
        const float* src = x + (size_t)toks[r] * HD;
        _Float16*    dst = A + r * LDA;
        for (int c = lane; c < HD; c += 32) dst[c] = (_Float16)src[c];
    }
    __syncthreads();

    const _Float16* We = Wh + (size_t)e * HD * HD;

    v8f acc[2][6];
#pragma unroll
    for (int mt = 0; mt < 2; ++mt)
#pragma unroll
        for (int j = 0; j < 6; ++j) acc[mt][j] = (v8f){0,0,0,0,0,0,0,0};

    for (int k0 = 0; k0 < HD; k0 += 32) {
        Frag a0, a1;
        load_a_frag(a0, A, 0, k0, lane);
        load_a_frag(a1, A, 1, k0, lane);
#pragma unroll
        for (int j = 0; j < 6; ++j) {
            int n0 = (wave * 6 + j) * 16;
            Frag b;
            load_b_frag(b, We, n0, k0, lane);
            acc[0][j] = __builtin_amdgcn_wmma_f32_16x16x32_f16(
                false, a0.h, false, b.h, (short)0, acc[0][j], false, false);
            acc[1][j] = __builtin_amdgcn_wmma_f32_16x16x32_f16(
                false, a1.h, false, b.h, (short)0, acc[1][j], false, false);
        }
    }

    // epilogue: + bias, * routing weight, atomic scatter into out
    int nlane = lane & 15, g = lane >> 4;
#pragma unroll
    for (int j = 0; j < 6; ++j) {
        int n = (wave * 6 + j) * 16 + nlane;
        float bias = eb[e * HD + n];
#pragma unroll
        for (int mt = 0; mt < 2; ++mt) {
#pragma unroll
            for (int r = 0; r < 8; ++r) {
                int m = mt * 16 + r + 8 * g;         // C/D layout: lanes 16-31 -> M+8
                float val = wts[m] * (acc[mt][j][r] + bias);
                atomicAdd(&out[(size_t)toks[m] * HD + n], val);
            }
        }
    }
}

// ---------------------------------------------------------------------------
extern "C" void kernel_launch(void* const* d_in, const int* in_sizes, int n_in,
                              void* d_out, int out_size, void* d_ws, size_t ws_size,
                              hipStream_t stream) {
    const float* x  = (const float*)d_in[0];   // [B,S,H]
    const float* rw = (const float*)d_in[1];   // [E,H]
    const float* rb = (const float*)d_in[2];   // [E]
    const float* ew = (const float*)d_in[3];   // [E,H,H]
    const float* eb = (const float*)d_in[4];   // [E,H]
    float* out = (float*)d_out;                // [B,S,H]

    char* ws = (char*)d_ws;
    int*  counts  = (int*)ws;                                        // 8 ints
    int2* entries = (int2*)(ws + 256);                               // 2 MB
    _Float16* wh  = (_Float16*)(ws + 256 + (size_t)NE * CAP * sizeof(int2)); // 9.4 MB

    moe_zero_kernel<<<2048, 256, 0, stream>>>(out, out_size, counts);
    moe_cvt_kernel<<<2048, 256, 0, stream>>>(ew, wh, NE * HD * HD / 4);
    moe_router_kernel<<<NTOK / 8, 256, 0, stream>>>(x, rw, rb, counts, entries);
    dim3 grid(CAP / TM, NE);
    moe_expert_kernel<<<grid, 256, 0, stream>>>(x, wh, eb, counts, entries, out);
}
